// UNetScoreNetwork_56831007261037
// MI455X (gfx1250) — matrix operand order repaired and checked by reading, not compiled, not run on hardware
//
#include <hip/hip_runtime.h>

typedef __attribute__((ext_vector_type(16))) __bf16       v16bf;
typedef __attribute__((ext_vector_type(8)))  __bf16       v8bf;
typedef __attribute__((ext_vector_type(8)))  float        v8f;
typedef __attribute__((ext_vector_type(8)))  unsigned int v8u;

#define Bsz  4
#define Npts 8192
#define Mpts 4096
#define XS   768   // xbuf row stride in bf16 elements (max concat width 512+256)
#define AS   512   // acc buffer row stride
#define MPTS 64    // points per workgroup (4 M-tiles of 16) -> 4x weight reuse

// ---------------------------------------------------------------------------
// 1) Nearest-neighbor search: per noisy point, argmin over 4096 targets.
//    Writes bf16 concat input [noisy(3) | nearest(3) | zero-pad to 32].
// ---------------------------------------------------------------------------
__global__ __launch_bounds__(256) void nn_kernel(const float* __restrict__ noisy,
                                                 const float* __restrict__ target,
                                                 __bf16* __restrict__ x0) {
  __shared__ float tt[512 * 3];
  const int gid = blockIdx.x * 256 + threadIdx.x;   // block stays within one batch
  const int b = gid / Npts;
  const float* p = noisy + (size_t)gid * 3;
  const float px = p[0], py = p[1], pz = p[2];
  float best = 3.4e38f, bx = 0.f, by = 0.f, bz = 0.f;
  for (int mt = 0; mt < Mpts / 512; ++mt) {
    __syncthreads();
    const float* tsrc = target + ((size_t)b * Mpts + mt * 512) * 3;
    for (int i = threadIdx.x; i < 512 * 3; i += 256) tt[i] = tsrc[i];
    __syncthreads();
#pragma unroll 4
    for (int q = 0; q < 512; ++q) {
      float tx = tt[q * 3 + 0], ty = tt[q * 3 + 1], tz = tt[q * 3 + 2];
      float dx = px - tx, dy = py - ty, dz = pz - tz;
      float d = dx * dx + dy * dy + dz * dz;
      if (d < best) { best = d; bx = tx; by = ty; bz = tz; }
    }
  }
  __bf16* row = x0 + (size_t)gid * 32;
  row[0] = (__bf16)px; row[1] = (__bf16)py; row[2] = (__bf16)pz;
  row[3] = (__bf16)bx; row[4] = (__bf16)by; row[5] = (__bf16)bz;
  for (int c = 6; c < 32; ++c) row[c] = (__bf16)0.f;
}

// ---------------------------------------------------------------------------
// 2) Time embedding: sinusoidal -> 128x512 silu -> 512x512. B=4, tiny fp32.
// ---------------------------------------------------------------------------
__global__ __launch_bounds__(256) void temb_kernel(const float* __restrict__ ts,
                                                   const float* __restrict__ w1,
                                                   const float* __restrict__ b1,
                                                   const float* __restrict__ w2,
                                                   const float* __restrict__ b2,
                                                   float* __restrict__ temb) {
  __shared__ float emb[4 * 128];
  __shared__ float h[4 * 512];
  const int t = threadIdx.x;
  {
    int b = t >> 6, i = t & 63;
    float freq = __expf((float)i * (-9.210340371976184f / 63.f)); // -ln(1e4)/(half-1)
    float v = ts[b] * freq;
    emb[b * 128 + i]      = sinf(v);
    emb[b * 128 + 64 + i] = cosf(v);
  }
  __syncthreads();
  for (int o = 0; o < 8; ++o) {
    int idx = t + o * 256;
    int b = idx >> 9, j = idx & 511;
    float s = b1[j];
    for (int k = 0; k < 128; ++k) s += emb[b * 128 + k] * w1[k * 512 + j];
    h[b * 512 + j] = s / (1.f + __expf(-s));   // silu
  }
  __syncthreads();
  for (int o = 0; o < 8; ++o) {
    int idx = t + o * 256;
    int b = idx >> 9, j = idx & 511;
    float s = b2[j];
    for (int k = 0; k < 512; ++k) s += h[b * 512 + k] * w2[k * 512 + j];
    temb[idx] = s;
  }
}

// ---------------------------------------------------------------------------
// 3) Fold per-batch temb through the bottom half of mid_w:
//    tbias[b][j] = mid_b[j] + sum_k temb[b][k] * mid_w[512+k][j]
// ---------------------------------------------------------------------------
__global__ __launch_bounds__(256) void tbias_kernel(const float* __restrict__ temb,
                                                    const float* __restrict__ mid_w,
                                                    const float* __restrict__ mid_b,
                                                    float* __restrict__ tbias) {
  int idx = blockIdx.x * 256 + threadIdx.x;   // 2048 = 4*512
  int b = idx >> 9, j = idx & 511;
  const float* e = temb + b * 512;
  const float* w = mid_w + (size_t)512 * 512 + j;
  float s = mid_b[j];
  for (int k = 0; k < 512; ++k) s += e[k] * w[(size_t)k * 512];
  tbias[idx] = s;
}

// ---------------------------------------------------------------------------
// 4) Weight transpose + bf16 convert: dst[n][k] = (k<din) ? src[k][n] : 0
// ---------------------------------------------------------------------------
__global__ __launch_bounds__(256) void wconv_kernel(const float* __restrict__ src,
                                                    __bf16* __restrict__ dst,
                                                    int din, int dout, int kpad) {
  int idx = blockIdx.x * 256 + threadIdx.x;
  if (idx >= dout * kpad) return;
  int n = idx / kpad, k = idx - n * kpad;
  dst[idx] = (k < din) ? (__bf16)src[(size_t)k * dout + n] : (__bf16)0.f;
}

// ---------------------------------------------------------------------------
// 5) Fused per-tile U-Net MLP. 64 points per workgroup, 8 waves.
// ---------------------------------------------------------------------------
struct FusedParams {
  const __bf16* x0;        // (B,N,32) bf16 padded concat input
  const float*  tbias;     // (B,512) folded mid bias
  const __bf16* wt[6];     // transposed bf16 weights [dout][Kpad]
  const float*  bias[6];   // slot 3 unused (tbias used instead)
  const float*  gamma[6];
  const float*  beta[6];
  const float*  out_w;     // (256,3) fp32
  const float*  out_b;     // (3)
  float*        out;       // (B,N,3) fp32
};

// A-operand gather: contiguous K order matching B, K = 16*g + 2*v + h
__device__ __forceinline__ v16bf load_a(const __bf16* in_lds, int row, int g, int kbase) {
  v8u au;
#pragma unroll
  for (int v = 0; v < 8; ++v) {
    int kk = kbase + (g << 4) + (v << 1);
    au[v] = *(const unsigned int*)(in_lds + row * XS + kk);
  }
  return __builtin_bit_cast(v16bf, au);
}

// One Linear(K->NOUT) + bias + LayerNorm + SiLU for a 64-point tile.
// Each B tile (weights, L2-resident) is reused across 4 A tiles (4 M-tiles).
template <int K, int NOUT>
__device__ __forceinline__ void layer_wmma(
    const __bf16* __restrict__ wt, const float* __restrict__ bias,
    const float* __restrict__ gamma, const float* __restrict__ beta,
    const __bf16* in_lds, __bf16* accb, float* red,
    __bf16* dst0, int s0, __bf16* dst1, int s1) {
  const int tid  = threadIdx.x;
  const int wave = tid >> 5;
  const int lane = tid & 31;
  const int mrow = lane & 15;       // A row / B column (N) index
  const int g    = lane >> 4;       // lane half-group
  constexpr int NTB = NOUT / 128;   // n-tile blocks per wave
  constexpr int KT  = K / 32;

  for (int ntb = 0; ntb < NTB; ++ntb) {
    const int nt = wave + ntb * 8;
    v8f acc0 = {}, acc1 = {}, acc2 = {}, acc3 = {};
    const __bf16* wrow = wt + ((size_t)(nt * 16 + mrow)) * K + g * 16;
#pragma unroll 4
    for (int kt = 0; kt < KT; ++kt) {
      if (kt + 1 < KT) __builtin_prefetch(wrow + ((kt + 1) << 5), 0, 0);
      // B-operand: lanes 0-15 -> K 0..15, lanes 16-31 -> K 16..31, N = lane&15
      v16bf bm = *(const v16bf*)(wrow + (kt << 5));
      v16bf a0 = load_a(in_lds,  0 + mrow, g, kt << 5);
      v16bf a1 = load_a(in_lds, 16 + mrow, g, kt << 5);
      v16bf a2 = load_a(in_lds, 32 + mrow, g, kt << 5);
      v16bf a3 = load_a(in_lds, 48 + mrow, g, kt << 5);
      acc0 = __builtin_amdgcn_wmma_f32_16x16x32_bf16(false, a0, false, bm, (short)0, acc0, false, false);
      acc1 = __builtin_amdgcn_wmma_f32_16x16x32_bf16(false, a1, false, bm, (short)0, acc1, false, false);
      acc2 = __builtin_amdgcn_wmma_f32_16x16x32_bf16(false, a2, false, bm, (short)0, acc2, false, false);
      acc3 = __builtin_amdgcn_wmma_f32_16x16x32_bf16(false, a3, false, bm, (short)0, acc3, false, false);
    }
    // C/D layout: VGPR r, lane -> row = r + 8*(lane/16), col = lane&15
    const int col = (nt << 4) + mrow;
    const float bcol = bias[col];
#pragma unroll
    for (int r = 0; r < 8; ++r) {
      int row = r + (g << 3);
      accb[(row +  0) * AS + col] = (__bf16)(acc0[r] + bcol);
      accb[(row + 16) * AS + col] = (__bf16)(acc1[r] + bcol);
      accb[(row + 32) * AS + col] = (__bf16)(acc2[r] + bcol);
      accb[(row + 48) * AS + col] = (__bf16)(acc3[r] + bcol);
    }
  }
  __syncthreads();

  // LayerNorm + SiLU: 4 threads per point, contiguous chunks, b128 LDS ops
  constexpr int CHUNK = NOUT / 4;
  const int pt = tid >> 2;
  const int j  = tid & 3;
  const __bf16* crow = accb + pt * AS + j * CHUNK;
  float s = 0.f, ss = 0.f;
#pragma unroll
  for (int i = 0; i < CHUNK / 8; ++i) {
    uint4 u = *(const uint4*)(crow + i * 8);
    v8bf h = __builtin_bit_cast(v8bf, u);
#pragma unroll
    for (int e = 0; e < 8; ++e) { float v = (float)h[e]; s += v; ss += v * v; }
  }
  red[tid] = s; red[256 + tid] = ss;
  __syncthreads();
  if (j == 0) {
    float ts = 0.f, tss = 0.f;
#pragma unroll
    for (int q = 0; q < 4; ++q) { ts += red[(pt << 2) + q]; tss += red[256 + (pt << 2) + q]; }
    float mean = ts / (float)NOUT;
    float var  = tss / (float)NOUT - mean * mean;
    red[512 + pt] = mean;
    red[576 + pt] = rsqrtf(var + 1e-5f);
  }
  __syncthreads();
  const float mean = red[512 + pt], rstd = red[576 + pt];
#pragma unroll
  for (int i = 0; i < CHUNK / 8; ++i) {
    uint4 u = *(const uint4*)(crow + i * 8);
    v8bf h = __builtin_bit_cast(v8bf, u);
    v8bf o;
#pragma unroll
    for (int e = 0; e < 8; ++e) {
      int c = j * CHUNK + i * 8 + e;
      float v = ((float)h[e] - mean) * rstd * gamma[c] + beta[c];
      o[e] = (__bf16)(v / (1.f + __expf(-v)));   // silu
    }
    uint4 ou = __builtin_bit_cast(uint4, o);
    *(uint4*)(dst0 + pt * s0 + j * CHUNK + i * 8) = ou;
    if (dst1) *(uint4*)(dst1 + pt * s1 + j * CHUNK + i * 8) = ou;
  }
  __syncthreads();
}

__global__ __launch_bounds__(256) void fused_unet(FusedParams p) {
  extern __shared__ char smem[];
  __bf16* xbuf  = (__bf16*)smem;             // 64 x 768  (96 KB)
  __bf16* accb  = xbuf  + MPTS * XS;         // 64 x 512  (64 KB)
  __bf16* skip0 = accb  + MPTS * AS;         // 64 x 128  (16 KB)
  __bf16* skip1 = skip0 + MPTS * 128;        // 64 x 256  (32 KB)
  float*  red   = (float*)(skip1 + MPTS * 256);  // 768 floats

  const int tid = threadIdx.x;
  const int b   = blockIdx.x / (Npts / MPTS);
  const int n0  = (blockIdx.x % (Npts / MPTS)) * MPTS;

  // stage 64x32 bf16 input tile: one uint4 (8 elems) per thread
  {
    const uint4* src = (const uint4*)(p.x0 + ((size_t)(b * Npts + n0)) * 32);
    int pt = tid >> 2, c = (tid & 3) * 8;
    *(uint4*)(xbuf + pt * XS + c) = src[tid];
  }
  __syncthreads();

  layer_wmma< 32, 128>(p.wt[0], p.bias[0], p.gamma[0], p.beta[0],
                       xbuf, accb, red, xbuf, XS, skip0, 128);
  layer_wmma<128, 256>(p.wt[1], p.bias[1], p.gamma[1], p.beta[1],
                       xbuf, accb, red, xbuf, XS, skip1, 256);
  layer_wmma<256, 512>(p.wt[2], p.bias[2], p.gamma[2], p.beta[2],
                       xbuf, accb, red, xbuf, XS, nullptr, 0);
  // mid with temb folded into per-batch bias -> K = 512
  layer_wmma<512, 512>(p.wt[3], p.tbias + b * 512, p.gamma[3], p.beta[3],
                       xbuf, accb, red, xbuf, XS, nullptr, 0);
  // append skip1 -> xbuf[:,512:768)
  {
    const uint4* s = (const uint4*)skip1;
#pragma unroll
    for (int i = tid; i < MPTS * 32; i += 256) {   // 32 uint4 per 256-elem row
      int pt = i >> 5, c = (i & 31) * 8;
      *(uint4*)(xbuf + pt * XS + 512 + c) = s[i];
    }
  }
  __syncthreads();
  layer_wmma<768, 512>(p.wt[4], p.bias[4], p.gamma[4], p.beta[4],
                       xbuf, accb, red, xbuf, XS, nullptr, 0);
  // append skip0 -> xbuf[:,512:640)
  {
    const uint4* s = (const uint4*)skip0;
#pragma unroll
    for (int i = tid; i < MPTS * 16; i += 256) {   // 16 uint4 per 128-elem row
      int pt = i >> 4, c = (i & 15) * 8;
      *(uint4*)(xbuf + pt * XS + 512 + c) = s[i];
    }
  }
  __syncthreads();
  layer_wmma<640, 256>(p.wt[5], p.bias[5], p.gamma[5], p.beta[5],
                       xbuf, accb, red, xbuf, XS, nullptr, 0);

  // output projection 256 -> 3 (fp32 VALU, 4 threads per point)
  {
    int pt = tid >> 2, j = tid & 3;
    float p0 = 0.f, p1 = 0.f, p2 = 0.f;
#pragma unroll
    for (int i = 0; i < 8; ++i) {
      uint4 u = *(const uint4*)(xbuf + pt * XS + j * 64 + i * 8);
      v8bf h = __builtin_bit_cast(v8bf, u);
#pragma unroll
      for (int e = 0; e < 8; ++e) {
        int k = j * 64 + i * 8 + e;
        float v = (float)h[e];
        p0 += v * p.out_w[k * 3 + 0];
        p1 += v * p.out_w[k * 3 + 1];
        p2 += v * p.out_w[k * 3 + 2];
      }
    }
    red[tid] = p0; red[256 + tid] = p1; red[512 + tid] = p2;
    __syncthreads();
    if (j == 0) {
      float o0 = p.out_b[0], o1 = p.out_b[1], o2 = p.out_b[2];
#pragma unroll
      for (int q = 0; q < 4; ++q) {
        o0 += red[(pt << 2) + q];
        o1 += red[256 + (pt << 2) + q];
        o2 += red[512 + (pt << 2) + q];
      }
      float* dst = p.out + ((size_t)(b * Npts + n0 + pt)) * 3;
      dst[0] = o0; dst[1] = o1; dst[2] = o2;
    }
  }
}

// ---------------------------------------------------------------------------
extern "C" void kernel_launch(void* const* d_in, const int* in_sizes, int n_in,
                              void* d_out, int out_size, void* d_ws, size_t ws_size,
                              hipStream_t stream) {
  (void)in_sizes; (void)n_in; (void)out_size; (void)ws_size;
  const float* noisy  = (const float*)d_in[0];
  const float* target = (const float*)d_in[1];
  const float* tsteps = (const float*)d_in[2];
  const float* te_w1  = (const float*)d_in[3];
  const float* te_b1  = (const float*)d_in[4];
  const float* te_w2  = (const float*)d_in[5];
  const float* te_b2  = (const float*)d_in[6];
  const float* enc_w[3]  = {(const float*)d_in[7],  (const float*)d_in[11], (const float*)d_in[15]};
  const float* enc_b[3]  = {(const float*)d_in[8],  (const float*)d_in[12], (const float*)d_in[16]};
  const float* enc_g[3]  = {(const float*)d_in[9],  (const float*)d_in[13], (const float*)d_in[17]};
  const float* enc_be[3] = {(const float*)d_in[10], (const float*)d_in[14], (const float*)d_in[18]};
  const float* mid_w  = (const float*)d_in[19];
  const float* mid_b  = (const float*)d_in[20];
  const float* mid_g  = (const float*)d_in[21];
  const float* mid_be = (const float*)d_in[22];
  const float* dec_w[2]  = {(const float*)d_in[23], (const float*)d_in[27]};
  const float* dec_b[2]  = {(const float*)d_in[24], (const float*)d_in[28]};
  const float* dec_g[2]  = {(const float*)d_in[25], (const float*)d_in[29]};
  const float* dec_be[2] = {(const float*)d_in[26], (const float*)d_in[30]};
  const float* out_w = (const float*)d_in[31];
  const float* out_b = (const float*)d_in[32];

  char* ws = (char*)d_ws;
  size_t off = 0;
  auto alloc = [&](size_t bytes) {
    char* q = ws + off; off += (bytes + 255) & ~(size_t)255; return q;
  };
  __bf16* x0    = (__bf16*)alloc((size_t)Bsz * Npts * 32 * 2);  // 2 MB
  float*  temb  = (float*) alloc((size_t)Bsz * 512 * 4);
  float*  tbias = (float*) alloc((size_t)Bsz * 512 * 4);
  __bf16* wt0   = (__bf16*)alloc((size_t)128 * 32  * 2);
  __bf16* wt1   = (__bf16*)alloc((size_t)256 * 128 * 2);
  __bf16* wt2   = (__bf16*)alloc((size_t)512 * 256 * 2);
  __bf16* wtm   = (__bf16*)alloc((size_t)512 * 512 * 2);
  __bf16* wtd0  = (__bf16*)alloc((size_t)512 * 768 * 2);
  __bf16* wtd1  = (__bf16*)alloc((size_t)256 * 640 * 2);

  nn_kernel<<<(Bsz * Npts) / 256, 256, 0, stream>>>(noisy, target, x0);
  temb_kernel<<<1, 256, 0, stream>>>(tsteps, te_w1, te_b1, te_w2, te_b2, temb);
  tbias_kernel<<<8, 256, 0, stream>>>(temb, mid_w, mid_b, tbias);
  wconv_kernel<<<(128 * 32)  / 256, 256, 0, stream>>>(enc_w[0], wt0, 6,   128, 32);
  wconv_kernel<<<(256 * 128) / 256, 256, 0, stream>>>(enc_w[1], wt1, 128, 256, 128);
  wconv_kernel<<<(512 * 256) / 256, 256, 0, stream>>>(enc_w[2], wt2, 256, 512, 256);
  wconv_kernel<<<(512 * 512) / 256, 256, 0, stream>>>(mid_w,    wtm, 512, 512, 512); // top 512 rows
  wconv_kernel<<<(512 * 768) / 256, 256, 0, stream>>>(dec_w[0], wtd0, 768, 512, 768);
  wconv_kernel<<<(256 * 640) / 256, 256, 0, stream>>>(dec_w[1], wtd1, 640, 256, 640);

  FusedParams fp;
  fp.x0 = x0; fp.tbias = tbias;
  fp.wt[0] = wt0; fp.wt[1] = wt1; fp.wt[2] = wt2;
  fp.wt[3] = wtm; fp.wt[4] = wtd0; fp.wt[5] = wtd1;
  fp.bias[0] = enc_b[0]; fp.bias[1] = enc_b[1]; fp.bias[2] = enc_b[2];
  fp.bias[3] = mid_b;    fp.bias[4] = dec_b[0]; fp.bias[5] = dec_b[1];
  fp.gamma[0] = enc_g[0]; fp.gamma[1] = enc_g[1]; fp.gamma[2] = enc_g[2];
  fp.gamma[3] = mid_g;    fp.gamma[4] = dec_g[0]; fp.gamma[5] = dec_g[1];
  fp.beta[0] = enc_be[0]; fp.beta[1] = enc_be[1]; fp.beta[2] = enc_be[2];
  fp.beta[3] = mid_be;    fp.beta[4] = dec_be[0]; fp.beta[5] = dec_be[1];
  fp.out_w = out_w; fp.out_b = out_b;
  fp.out = (float*)d_out;

  // dynamic LDS: 64*(768+512+128+256)*2 B activations + 768 floats reductions
  constexpr size_t SMEM = (size_t)MPTS * (XS + AS + 128 + 256) * 2 + 768 * 4; // 216,064 B
  fused_unet<<<(Bsz * Npts) / MPTS, 256, SMEM, stream>>>(fp);
}
